// MoE_87514253623549
// MI455X (gfx1250) — matrix operand order repaired and checked by reading, not compiled, not run on hardware
//
#include <hip/hip_runtime.h>

typedef __bf16 bf16;
typedef __attribute__((ext_vector_type(16))) __bf16 v16bf;
typedef __attribute__((ext_vector_type(4)))  __bf16 v4bf;
typedef __attribute__((ext_vector_type(8)))  float   v8f;

#define T_TOK 16384   // B*S
#define DDIM  512
#define FDIM  2048
#define NEXP  8
#define M_TILE 32
#define HS_STRIDE 264 // 256 + 8 pad -> row step = 4 LDS banks, conflict-free b128 reads

// ---- WMMA fragment loaders (contiguous K order shared by A and B, not the ISA table order) ----
// A 16x32 bf16: lane holds row M=lane&15, K = (lane>>4)*16 + 0..15 contiguous
__device__ __forceinline__ v16bf load_fragA(const bf16* rowbase, int hi) {
  union { uint4 q[2]; v16bf v; } u;
  u.q[0] = *(const uint4*)(rowbase + hi * 16);
  u.q[1] = *(const uint4*)(rowbase + hi * 16 + 8);
  return u.v;
}
// B 32x16 bf16: lane holds column N=lane&15, K = (lane>>4)*16 + 0..15 contiguous
__device__ __forceinline__ v16bf load_fragB(const bf16* rowbase, int hi) {
  union { uint4 q[2]; v16bf v; } u;
  const bf16* p = rowbase + hi * 16;
  u.q[0] = *(const uint4*)(p);
  u.q[1] = *(const uint4*)(p + 8);
  return u.v;
}

#define WMMA_BF16(A, B, C) \
  __builtin_amdgcn_wmma_f32_16x16x32_bf16(false, (A), false, (B), (short)0, (C), false, false)

// ---------------- router: logits, softmax, top-2, and x -> bf16 ----------------
__global__ void __launch_bounds__(256)
router_kernel(const float* __restrict__ x, const float* __restrict__ gw,
              bf16* __restrict__ xb, int* __restrict__ sel0, int* __restrict__ sel1,
              float* __restrict__ rw0, float* __restrict__ rw1)
{
  const int lane = threadIdx.x & 31;
  const int wid  = threadIdx.x >> 5;
  const int t    = blockIdx.x * 8 + wid;   // one wave per token

  float xv[16];
  {
    const float4* xr = (const float4*)(x + (size_t)t * DDIM + lane * 16);
#pragma unroll
    for (int i = 0; i < 4; ++i) {
      float4 f = xr[i];
      xv[4*i+0] = f.x; xv[4*i+1] = f.y; xv[4*i+2] = f.z; xv[4*i+3] = f.w;
    }
  }
  v16bf xvb;
#pragma unroll
  for (int i = 0; i < 16; ++i) xvb[i] = (bf16)xv[i];
  *(v16bf*)(xb + (size_t)t * DDIM + lane * 16) = xvb;

  float lg[8];
#pragma unroll
  for (int e = 0; e < NEXP; ++e) {
    const float4* gr = (const float4*)(gw + e * DDIM + lane * 16);
    float s = 0.0f;
#pragma unroll
    for (int i = 0; i < 4; ++i) {
      float4 g = gr[i];
      s += xv[4*i+0]*g.x + xv[4*i+1]*g.y + xv[4*i+2]*g.z + xv[4*i+3]*g.w;
    }
#pragma unroll
    for (int off = 16; off >= 1; off >>= 1) s += __shfl_xor(s, off, 32);
    lg[e] = s;
  }
  float mx = lg[0];
#pragma unroll
  for (int e = 1; e < NEXP; ++e) mx = fmaxf(mx, lg[e]);
  float p[8];
#pragma unroll
  for (int e = 0; e < NEXP; ++e) p[e] = __expf(lg[e] - mx);
  int e0 = 0; float b0 = p[0];
#pragma unroll
  for (int e = 1; e < NEXP; ++e) if (p[e] > b0) { b0 = p[e]; e0 = e; }
  int e1 = (e0 == 0) ? 1 : 0; float b1 = p[e1];
#pragma unroll
  for (int e = 0; e < NEXP; ++e) if (e != e0 && p[e] > b1) { b1 = p[e]; e1 = e; }
  const float denom = b0 + b1;   // softmax denominator cancels in renormalization
  if (lane == 0) {
    sel0[t] = e0; sel1[t] = e1;
    rw0[t] = b0 / denom; rw1[t] = b1 / denom;
  }
}

// ---------------- deterministic per-expert token list compaction ----------------
__global__ void __launch_bounds__(256)
build_lists_kernel(const int* __restrict__ sel0, const int* __restrict__ sel1,
                   const float* __restrict__ rw0, const float* __restrict__ rw1,
                   int* __restrict__ lists, float* __restrict__ wlists,
                   int* __restrict__ counts)
{
  const int e    = blockIdx.x;
  const int tid  = threadIdx.x;
  const int lane = tid & 31;
  const int wid  = tid >> 5;
  __shared__ int wtot[8];
  int running = 0;
  for (int base = 0; base < T_TOK; base += 256) {
    const int tok = base + tid;
    const bool p0 = (sel0[tok] == e);
    const bool p1 = (sel1[tok] == e);
    const bool p  = p0 || p1;
    const float w = p0 ? rw0[tok] : rw1[tok];
    unsigned long long m = __ballot(p);
    const int pre = __popcll(m & ((1ull << lane) - 1ull));
    if (lane == 0) wtot[wid] = __popcll(m);
    __syncthreads();
    int below = 0, tot = 0;
#pragma unroll
    for (int wdx = 0; wdx < 8; ++wdx) { int v = wtot[wdx]; tot += v; if (wdx < wid) below += v; }
    if (p) {
      const int pos = running + below + pre;
      lists[e * T_TOK + pos]  = tok;
      wlists[e * T_TOK + pos] = w;
    }
    running += tot;
    __syncthreads();
  }
  if (tid == 0) counts[e] = running;
}

// ---------------- weight fp32 -> bf16 conversion (one pass) ----------------
__global__ void __launch_bounds__(256)
cvt_weights_kernel(const float* __restrict__ w1, const float* __restrict__ w2,
                   const float* __restrict__ w3,
                   bf16* __restrict__ w1b, bf16* __restrict__ w2b, bf16* __restrict__ w3b)
{
  const size_t i = ((size_t)blockIdx.x * 256 + threadIdx.x) * 4;
  float4 a = *(const float4*)(w1 + i);
  float4 b = *(const float4*)(w2 + i);
  float4 c = *(const float4*)(w3 + i);
  v4bf oa, ob, oc;
  oa[0]=(bf16)a.x; oa[1]=(bf16)a.y; oa[2]=(bf16)a.z; oa[3]=(bf16)a.w;
  ob[0]=(bf16)b.x; ob[1]=(bf16)b.y; ob[2]=(bf16)b.z; ob[3]=(bf16)b.w;
  oc[0]=(bf16)c.x; oc[1]=(bf16)c.y; oc[2]=(bf16)c.z; oc[3]=(bf16)c.w;
  *(v4bf*)(w1b + i) = oa;
  *(v4bf*)(w2b + i) = ob;
  *(v4bf*)(w3b + i) = oc;
}

__global__ void __launch_bounds__(256)
zero_out_kernel(float* __restrict__ out)
{
  const size_t i = ((size_t)blockIdx.x * 256 + threadIdx.x) * 4;
  float4 z; z.x = z.y = z.z = z.w = 0.0f;
  *(float4*)(out + i) = z;
}

// ---------------- fused gathered expert FFN: relu(XW1^T)*(XW3^T) -> *W2^T ----------------
// Block = 8 waves, 32 gathered token rows. Each wave holds two 16-row A fragments and
// reuses every B fragment twice -> 12 b128 loads per 8 WMMAs.
__global__ void __launch_bounds__(256, 1)
moe_expert_kernel(const bf16* __restrict__ xb,
                  const bf16* __restrict__ w1b,
                  const bf16* __restrict__ w2b,
                  const bf16* __restrict__ w3b,
                  const int*  __restrict__ lists,
                  const float* __restrict__ wlists,
                  const int*  __restrict__ counts,
                  float* __restrict__ out)
{
  const int e    = blockIdx.y;
  const int tile = blockIdx.x;
  const int n_e  = counts[e];
  const int row0 = tile * M_TILE;
  if (row0 >= n_e) return;   // uniform exit: remaining waves keep EXEC all-ones for WMMA

  const int tid  = threadIdx.x;
  const int lane = tid & 31;
  const int wid  = tid >> 5;
  const int hi   = lane >> 4;
  const int ln   = lane & 15;

  __shared__ bf16 Hs[M_TILE * HS_STRIDE] __attribute__((aligned(32)));
  __shared__ int   tokS[M_TILE];
  __shared__ float wS[M_TILE];

  if (tid < M_TILE) {
    const int r = row0 + tid;
    const int safe = (r < n_e) ? r : row0;          // clamp OOB rows to a valid token
    tokS[tid] = lists[e * T_TOK + safe];
    wS[tid]   = (r < n_e) ? wlists[e * T_TOK + r] : 0.0f;  // weight 0 => zero contribution
  }
  __syncthreads();

  const bf16* w1e = w1b + (size_t)e * FDIM * DDIM;
  const bf16* w3e = w3b + (size_t)e * FDIM * DDIM;
  const bf16* w2e = w2b + (size_t)e * DDIM * FDIM;
  const bf16* xrow0 = xb + (size_t)tokS[ln]      * DDIM;  // gathered A rows (sub-tile 0)
  const bf16* xrow1 = xb + (size_t)tokS[16 + ln] * DDIM;  // gathered A rows (sub-tile 1)

  v8f accO[2][4] = {};         // [m][j]: 32 rows x 64 out-cols (cols wid*64 .. +63 of D)

  for (int fc = 0; fc < FDIM / 256; ++fc) {
    const int fbase = fc * 256;

    // ---- phase A: H[32,256] = relu(X W1^T) * (X W3^T) for this F-chunk ----
    v8f acc1[2][2] = {};
    v8f acc3[2][2] = {};
    const int nA = wid * 32;
#pragma unroll 2
    for (int k = 0; k < DDIM; k += 32) {
      v16bf a0 = load_fragA(xrow0 + k, hi);
      v16bf a1 = load_fragA(xrow1 + k, hi);
#pragma unroll
      for (int j = 0; j < 2; ++j) {
        const int f = fbase + nA + j * 16 + ln;
        v16bf b1 = load_fragB(w1e + (size_t)f * DDIM + k, hi);
        acc1[0][j] = WMMA_BF16(a0, b1, acc1[0][j]);
        acc1[1][j] = WMMA_BF16(a1, b1, acc1[1][j]);
        v16bf b3 = load_fragB(w3e + (size_t)f * DDIM + k, hi);
        acc3[0][j] = WMMA_BF16(a0, b3, acc3[0][j]);
        acc3[1][j] = WMMA_BF16(a1, b3, acc3[1][j]);
      }
    }
    __syncthreads();  // previous chunk's phase-B reads of Hs are done
#pragma unroll
    for (int m = 0; m < 2; ++m) {
#pragma unroll
      for (int j = 0; j < 2; ++j) {
        const int col = nA + j * 16 + ln;
#pragma unroll
        for (int r = 0; r < 8; ++r) {
          const int M = m * 16 + r + 8 * hi;         // C layout: M = r + 8*(lane>=16)
          float g = acc1[m][j][r];
          g = g > 0.0f ? g : 0.0f;
          const float h = g * acc3[m][j][r] * wS[M]; // fold routing weight into H
          Hs[M * HS_STRIDE + col] = (bf16)h;
        }
      }
    }
    __syncthreads();

    // ---- phase B: accO += H[32,256] @ W2[:, fbase:fbase+256]^T ----
    const int nB = wid * 64;
#pragma unroll 2
    for (int kk = 0; kk < 256; kk += 32) {
      v16bf a0 = load_fragA(&Hs[ln * HS_STRIDE + kk], hi);
      v16bf a1 = load_fragA(&Hs[(16 + ln) * HS_STRIDE + kk], hi);
#pragma unroll
      for (int j = 0; j < 4; ++j) {
        const int d = nB + j * 16 + ln;
        v16bf b = load_fragB(w2e + (size_t)d * FDIM + fbase + kk, hi);
        accO[0][j] = WMMA_BF16(a0, b, accO[0][j]);
        accO[1][j] = WMMA_BF16(a1, b, accO[1][j]);
      }
    }
  }

  // ---- scatter-add (exactly 2 expert contributions per element; fp add commutative) ----
  const int nB = wid * 64;
#pragma unroll
  for (int m = 0; m < 2; ++m) {
#pragma unroll
    for (int j = 0; j < 4; ++j) {
      const int col = nB + j * 16 + ln;
#pragma unroll
      for (int r = 0; r < 8; ++r) {
        const int M = m * 16 + r + 8 * hi;
        if (row0 + M < n_e) {
          atomicAdd(out + (size_t)tokS[M] * DDIM + col, accO[m][j][r]);
        }
      }
    }
  }
}

extern "C" void kernel_launch(void* const* d_in, const int* in_sizes, int n_in,
                              void* d_out, int out_size, void* d_ws, size_t ws_size,
                              hipStream_t stream)
{
  (void)in_sizes; (void)n_in; (void)out_size; (void)ws_size;
  const float* x  = (const float*)d_in[0];
  const float* gw = (const float*)d_in[1];
  const float* w1 = (const float*)d_in[2];
  const float* w2 = (const float*)d_in[3];   // note: dict order is x, gate_w, w1, w2, w3
  const float* w3 = (const float*)d_in[4];
  float* out = (float*)d_out;

  char* ws = (char*)d_ws;
  size_t off = 0;
  bf16* xb   = (bf16*)(ws + off); off += (size_t)T_TOK * DDIM * 2;          // 16 MB
  bf16* w1b  = (bf16*)(ws + off); off += (size_t)NEXP * FDIM * DDIM * 2;    // 16 MB
  bf16* w3b  = (bf16*)(ws + off); off += (size_t)NEXP * FDIM * DDIM * 2;    // 16 MB
  bf16* w2b  = (bf16*)(ws + off); off += (size_t)NEXP * DDIM * FDIM * 2;    // 16 MB
  int*   lists  = (int*)(ws + off);   off += (size_t)NEXP * T_TOK * 4;      // 512 KB
  float* wlists = (float*)(ws + off); off += (size_t)NEXP * T_TOK * 4;      // 512 KB
  int*   sel0 = (int*)(ws + off);   off += (size_t)T_TOK * 4;
  int*   sel1 = (int*)(ws + off);   off += (size_t)T_TOK * 4;
  float* rw0  = (float*)(ws + off); off += (size_t)T_TOK * 4;
  float* rw1  = (float*)(ws + off); off += (size_t)T_TOK * 4;
  int*   counts = (int*)(ws + off); off += 256;

  // 1) weights fp32 -> bf16 (E*F*D = 8388608 elems, 4/thread)
  cvt_weights_kernel<<<8192, 256, 0, stream>>>(w1, w2, w3, w1b, w2b, w3b);
  // 2) router + x -> bf16 (one wave per token)
  router_kernel<<<T_TOK / 8, 256, 0, stream>>>(x, gw, xb, sel0, sel1, rw0, rw1);
  // 3) deterministic expert token lists
  build_lists_kernel<<<NEXP, 256, 0, stream>>>(sel0, sel1, rw0, rw1, lists, wlists, counts);
  // 4) zero output (T*D = 8388608 floats)
  zero_out_kernel<<<8192, 256, 0, stream>>>(out);
  // 5) fused gathered expert FFN; over-launched tiles early-exit on counts[e]
  moe_expert_kernel<<<dim3(T_TOK / M_TILE, NEXP), 256, 0, stream>>>(xb, w1b, w2b, w3b,
                                                                    lists, wlists, counts, out);
}
